// DHCN_44873818308693
// MI455X (gfx1250) — compile-verified, matrix operand order checked by reading, not run on hardware
//
#include <hip/hip_runtime.h>
#include <hip/hip_bf16.h>

#define N_NODE 100000
#define EMB    100
#define BATCH  512
#define SEQ    50
#define NNZ    1000000

typedef __attribute__((ext_vector_type(16))) _Float16 v16h;
typedef __attribute__((ext_vector_type(8)))  _Float16 v8h;
typedef __attribute__((ext_vector_type(8)))  float    v8f;

// padded GEMM geometry
#define K1P 224   // concat K: 200 -> 224
#define NP  112   // N: 100 -> 112
#define K2P 128   // glu K: 100 -> 128

// ---------------- fast-math helpers (hardware transcendentals) ----------------
__device__ __forceinline__ float fast_rcp(float x) { return __builtin_amdgcn_rcpf(x); }
__device__ __forceinline__ float fast_sigmoid(float v) {
  return fast_rcp(1.f + __expf(-v));
}
__device__ __forceinline__ float fast_tanh(float x) {
  float t = __expf(-2.f * fabsf(x));             // in (0,1], no overflow
  float y = (1.f - t) * fast_rcp(1.f + t);
  return copysignf(y, x);
}

// ---------------- vectorized elementwise helpers (float4 = b128) ----------------
__global__ void fill_zero_f32_v4(float4* p, int n4) {
  int i = blockIdx.x * blockDim.x + threadIdx.x;
  if (i < n4) p[i] = make_float4(0.f, 0.f, 0.f, 0.f);
}
__global__ void copy_f32_v4(float4* dst, const float4* src, int n4) {
  int i = blockIdx.x * blockDim.x + threadIdx.x;
  if (i < n4) dst[i] = src[i];
}
__global__ void add_f32_v4(float4* dst, const float4* src, int n4) {
  int i = blockIdx.x * blockDim.x + threadIdx.x;
  if (i < n4) {
    float4 a = dst[i], b = src[i];
    dst[i] = make_float4(a.x + b.x, a.y + b.y, a.z + b.z, a.w + b.w);
  }
}

// ---------------- HyperConv SpMM (COO, atomic scatter-add) ----------------
// one wave per edge; lanes stripe the 100-wide embedding row
__global__ void spmm_coo(const int* __restrict__ rows, const int* __restrict__ cols,
                         const float* __restrict__ vals,
                         const float* __restrict__ hin, float* __restrict__ hout) {
  int e = blockIdx.x * (blockDim.x >> 5) + (threadIdx.x >> 5);
  int lane = threadIdx.x & 31;
  if (e >= NNZ) return;
  int r = rows[e], c = cols[e];
  float v = vals[e];
  const float* src = hin + c * EMB;
  float* dst = hout + r * EMB;
  for (int d = lane; d < EMB; d += 32)
    atomicAdd(&dst[d], v * src[d]);
}

// ---------------- gathers / means ----------------
// b128 row gather: thread per (row, 4-float group); EMB row base = 400B aligned
__global__ void gather_seq_v4(const int* __restrict__ rev, const float* __restrict__ item_hg,
                              float* __restrict__ seq_h) {
  int i = blockIdx.x * blockDim.x + threadIdx.x;   // BATCH*SEQ*(EMB/4)
  if (i >= BATCH * SEQ * (EMB / 4)) return;
  int d4 = i % (EMB / 4);
  int rl = i / (EMB / 4);
  int idx = rev[rl];
  float4 v = make_float4(0.f, 0.f, 0.f, 0.f);
  if (idx != 0) v = *(const float4*)(item_hg + (idx - 1) * EMB + d4 * 4);
  *(float4*)(seq_h + rl * EMB + d4 * 4) = v;
}

__global__ void mean_seq(const float* __restrict__ seq_h, const float* __restrict__ slen,
                         float* __restrict__ hs) {
  int i = blockIdx.x * blockDim.x + threadIdx.x;   // BATCH*EMB
  if (i >= BATCH * EMB) return;
  int b = i / EMB, d = i % EMB;
  float a = 0.f;
  for (int l = 0; l < SEQ; ++l) a += seq_h[(b * SEQ + l) * EMB + d];
  hs[i] = a * fast_rcp(slen[b]);
}

__global__ void s0_kernel(const int* __restrict__ sess, const float* __restrict__ emb,
                          const float* __restrict__ slen,
                          float* __restrict__ s0, float* __restrict__ sess_lg) {
  int i = blockIdx.x * blockDim.x + threadIdx.x;   // BATCH*EMB
  if (i >= BATCH * EMB) return;
  int b = i / EMB, d = i % EMB;
  float a = 0.f;
  for (int l = 0; l < SEQ; ++l) {
    int idx = sess[b * SEQ + l];
    if (idx != 0) a += emb[(idx - 1) * EMB + d];
  }
  a *= fast_rcp(slen[b]);
  s0[i] = a;
  sess_lg[i] = a;
}

// ---------------- f16 pack kernels (pad + transpose + convert) ----------------
__global__ void pack_X(const float* __restrict__ pos_w, const float* __restrict__ seq_h,
                       _Float16* __restrict__ Xh) {
  int i = blockIdx.x * blockDim.x + threadIdx.x;   // 25600*K1P
  if (i >= BATCH * SEQ * K1P) return;
  int k = i % K1P, r = i / K1P;
  float v = 0.f;
  if (k < EMB) v = pos_w[(r % SEQ) * EMB + k];
  else if (k < 2 * EMB) v = seq_h[r * EMB + (k - EMB)];
  Xh[i] = (_Float16)v;
}
__global__ void pack_w1t(const float* __restrict__ w1, _Float16* __restrict__ W1t) {
  int i = blockIdx.x * blockDim.x + threadIdx.x;   // NP*K1P
  if (i >= NP * K1P) return;
  int k = i % K1P, n = i / K1P;
  W1t[i] = (n < EMB && k < 2 * EMB) ? (_Float16)w1[k * EMB + n] : (_Float16)0.f;
}
__global__ void pack_g1t(const float* __restrict__ g1, _Float16* __restrict__ G1t) {
  int i = blockIdx.x * blockDim.x + threadIdx.x;   // NP*K2P
  if (i >= NP * K2P) return;
  int k = i % K2P, n = i / K2P;
  G1t[i] = (n < EMB && k < EMB) ? (_Float16)g1[n * EMB + k] : (_Float16)0.f;
}
__global__ void pack_f16(const float* __restrict__ src, _Float16* __restrict__ dst, int n) {
  int i = blockIdx.x * blockDim.x + threadIdx.x;
  if (i < n) dst[i] = (_Float16)src[i];
}
__global__ void pack_t_f16(const float* __restrict__ src, _Float16* __restrict__ dst) {
  int i = blockIdx.x * blockDim.x + threadIdx.x;   // BATCH*BATCH
  if (i >= BATCH * BATCH) return;
  int k = i % BATCH, n = i / BATCH;
  dst[i] = (_Float16)src[k * BATCH + n];
}
__global__ void fill_zero_f16(_Float16* p, int n) {
  int i = blockIdx.x * blockDim.x + threadIdx.x;
  if (i < n) p[i] = (_Float16)0.f;
}

// ---------------- WMMA fragment loader ----------------
// per ISA 16-bit A/B layout: lane holds K runs [koff..koff+7] and [koff+16..koff+23]
// with koff = (lane>>4)*8 -> two aligned 16B loads per fragment
__device__ __forceinline__ v16h frag_ld(const _Float16* base, int koff) {
  v8h lo = *(const v8h*)(base + koff);
  v8h hi = *(const v8h*)(base + 16 + koff);
  v16h f;
#pragma unroll
  for (int i = 0; i < 8; ++i) { f[i] = lo[i]; f[8 + i] = hi[i]; }
  return f;
}

// ---------------- GEMM1: nh0h = f16(tanh(X @ w1)), [25600 x K2P] padded ----------------
__global__ void gemm1_wmma(const _Float16* __restrict__ Xh, const _Float16* __restrict__ W1t,
                           _Float16* __restrict__ nh0h) {
  int nt = blockIdx.x, mt = blockIdx.y;
  int lane = threadIdx.x;                 // wave32
  int m = mt * 16 + (lane & 15);
  int n = nt * 16 + (lane & 15);
  int koff = (lane >> 4) * 8;
  const _Float16* arow = Xh + m * K1P;
  const _Float16* brow = W1t + n * K1P;
  v8f acc = {};
#pragma unroll
  for (int kc = 0; kc < K1P / 32; ++kc) {
    v16h a = frag_ld(arow + kc * 32, koff);
    v16h b = frag_ld(brow + kc * 32, koff);
    acc = __builtin_amdgcn_wmma_f32_16x16x32_f16(false, a, false, b, (short)0, acc, false, false);
  }
#pragma unroll
  for (int r = 0; r < 8; ++r) {
    int row = mt * 16 + ((lane < 16) ? r : r + 8);
    nh0h[row * K2P + n] = (_Float16)fast_tanh(acc[r]);  // n < NP always; pad cols pre-zeroed
  }
}

// ---------------- GEMM2: nh = sigmoid(nh0 @ glu1_w^T + glu1_b + hsg[b]) ----------------
__global__ void gemm2_wmma(const _Float16* __restrict__ nh0h, const _Float16* __restrict__ G1t,
                           const float* __restrict__ glu1_b, const float* __restrict__ hsg,
                           float* __restrict__ nh) {
  int nt = blockIdx.x, mt = blockIdx.y;
  int lane = threadIdx.x;
  int m = mt * 16 + (lane & 15);
  int n = nt * 16 + (lane & 15);
  int koff = (lane >> 4) * 8;
  const _Float16* arow = nh0h + m * K2P;
  const _Float16* brow = G1t + n * K2P;
  v8f acc = {};
#pragma unroll
  for (int kc = 0; kc < K2P / 32; ++kc) {
    v16h a = frag_ld(arow + kc * 32, koff);
    v16h b = frag_ld(brow + kc * 32, koff);
    acc = __builtin_amdgcn_wmma_f32_16x16x32_f16(false, a, false, b, (short)0, acc, false, false);
  }
#pragma unroll
  for (int r = 0; r < 8; ++r) {
    int row = mt * 16 + ((lane < 16) ? r : r + 8);
    if (n < EMB) {
      float v = acc[r] + glu1_b[n] + hsg[(row / SEQ) * EMB + n];
      nh[row * EMB + n] = fast_sigmoid(v);
    }
  }
}

// ---------------- GEMM3: DA = D @ A  (512^3) ----------------
__global__ void gemmDA_wmma(const _Float16* __restrict__ Dh, const _Float16* __restrict__ Ath,
                            float* __restrict__ DA) {
  int nt = blockIdx.x, mt = blockIdx.y;
  int lane = threadIdx.x;
  int m = mt * 16 + (lane & 15);
  int n = nt * 16 + (lane & 15);
  int koff = (lane >> 4) * 8;
  const _Float16* arow = Dh + m * BATCH;
  const _Float16* brow = Ath + n * BATCH;
  v8f acc = {};
#pragma unroll
  for (int kc = 0; kc < BATCH / 32; ++kc) {
    v16h a = frag_ld(arow + kc * 32, koff);
    v16h b = frag_ld(brow + kc * 32, koff);
    acc = __builtin_amdgcn_wmma_f32_16x16x32_f16(false, a, false, b, (short)0, acc, false, false);
  }
#pragma unroll
  for (int r = 0; r < 8; ++r) {
    int row = mt * 16 + ((lane < 16) ? r : r + 8);
    DA[row * BATCH + n] = acc[r];
  }
}

// ---------------- small dense pieces ----------------
__global__ void hsg_kernel(const float* __restrict__ hs, const float* __restrict__ glu2_w,
                           float* __restrict__ hsg) {
  int i = blockIdx.x * blockDim.x + threadIdx.x;   // BATCH*EMB
  if (i >= BATCH * EMB) return;
  int b = i / EMB, j = i % EMB;
  float a = 0.f;
  for (int k = 0; k < EMB; ++k) a += hs[b * EMB + k] * glu2_w[j * EMB + k];
  hsg[i] = a;
}

__global__ void att_kernel(const float* __restrict__ nh, const float* __restrict__ w2,
                           const float* __restrict__ mask, float* __restrict__ att) {
  int r = blockIdx.x * blockDim.x + threadIdx.x;   // BATCH*SEQ
  if (r >= BATCH * SEQ) return;
  float a = 0.f;
  for (int j = 0; j < EMB; ++j) a += nh[r * EMB + j] * w2[j];
  att[r] = a * mask[r];
}

__global__ void sesshg_kernel(const float* __restrict__ att, const float* __restrict__ seq_h,
                              float* __restrict__ sess_hg) {
  int b = blockIdx.x;
  int d = threadIdx.x;
  if (d >= EMB) return;
  float a = 0.f;
  for (int l = 0; l < SEQ; ++l)
    a += att[b * SEQ + l] * seq_h[(b * SEQ + l) * EMB + d];
  sess_hg[b * EMB + d] = a;
}

__global__ void sprop_kernel(const float* __restrict__ DA, const float* __restrict__ s_in,
                             float* __restrict__ s_out, float* __restrict__ sess_lg) {
  int i = blockIdx.x * blockDim.x + threadIdx.x;   // BATCH*EMB
  if (i >= BATCH * EMB) return;
  int b = i / EMB, d = i % EMB;
  float a = 0.f;
  for (int k = 0; k < BATCH; ++k) a += DA[b * BATCH + k] * s_in[k * EMB + d];
  s_out[i] = a;
  sess_lg[i] += a;
}

// ---------------- SSL loss ----------------
__global__ void ssl_kernel(const float* __restrict__ sess_hg, const float* __restrict__ sess_lg,
                           const int* __restrict__ perm_rows, const int* __restrict__ perm_cols,
                           float* __restrict__ out) {
  __shared__ float red[BATCH];
  int b = threadIdx.x;
  int pr = perm_rows[b];
  float ps = 0.f, ns = 0.f;
  for (int d = 0; d < EMB; ++d) {
    ps += sess_hg[b * EMB + d] * sess_lg[b * EMB + d];
    ns += sess_lg[b * EMB + d] * sess_hg[pr * EMB + perm_cols[d]];
  }
  float sp = fast_sigmoid(ps);
  float sn = fast_sigmoid(ns);
  red[b] = -__logf(1e-8f + sp) - __logf(1e-8f + (1.f - sn));
  __syncthreads();
  for (int s = BATCH / 2; s > 0; s >>= 1) {
    if (b < s) red[b] += red[b + s];
    __syncthreads();
  }
  if (b == 0) out[0] = 0.01f * red[0];
}

// ---------------- host launcher ----------------
extern "C" void kernel_launch(void* const* d_in, const int* in_sizes, int n_in,
                              void* d_out, int out_size, void* d_ws, size_t ws_size,
                              hipStream_t stream) {
  const int*   session_item = (const int*)  d_in[0];
  const float* session_len  = (const float*)d_in[1];
  const float* D            = (const float*)d_in[2];
  const float* A            = (const float*)d_in[3];
  const int*   rev          = (const int*)  d_in[4];
  const float* mask         = (const float*)d_in[5];
  const float* emb          = (const float*)d_in[6];
  const float* pos_w        = (const float*)d_in[7];
  const float* w1           = (const float*)d_in[8];
  const float* w2           = (const float*)d_in[9];
  const float* glu1_w       = (const float*)d_in[10];
  const float* glu1_b       = (const float*)d_in[11];
  const float* glu2_w       = (const float*)d_in[12];
  const int*   adj_rows     = (const int*)  d_in[13];
  const int*   adj_cols     = (const int*)  d_in[14];
  const float* adj_vals     = (const float*)d_in[15];
  const int*   perm_rows    = (const int*)  d_in[16];
  const int*   perm_cols    = (const int*)  d_in[17];

  const int NE = N_NODE * EMB;                 // 10,000,000
  const int RM = BATCH * SEQ;                  // 25,600
  float* out     = (float*)d_out;
  float* item_hg = out;                        // [N_NODE, EMB]
  float* sess_hg = out + NE;                   // [BATCH, EMB]
  float* con     = sess_hg + BATCH * EMB;      // scalar

  float* ws  = (float*)d_ws;
  float* h_b = ws;                             // 10,000,000 (layer-0 output)
  float* h_a = ws + NE;                        // 10,000,000 (layer-1 output)

  // post-SpMM buffers reuse the h region (dead after item_hg finalized).
  // all offsets are multiples of 4 floats -> 16B alignment for b128 access.
  float*    seq_h   = ws;                                    //  2,560,000 f32
  _Float16* Xh      = (_Float16*)(ws + 2560000);             // 25600*224 f16
  _Float16* W1t     = (_Float16*)(ws + 5427200);             // 112*224   f16
  _Float16* nh0h    = (_Float16*)(ws + 5439744);             // 25600*128 f16
  _Float16* G1t     = (_Float16*)(ws + 7078144);             // 112*128   f16
  float*    nh      = ws + 7085312;                          //  2,560,000 f32
  float*    hs      = ws + 9645312;                          //     51,200
  float*    hsg     = ws + 9696512;                          //     51,200
  float*    att     = ws + 9747712;                          //     25,600
  _Float16* Dh      = (_Float16*)(ws + 9773312);             // 512*512 f16
  _Float16* Ath     = (_Float16*)(ws + 9904384);             // 512*512 f16
  float*    DA      = ws + 10035456;                         //    262,144
  float*    s_a     = ws + 10297600;                         //     51,200
  float*    s_b     = ws + 10348800;                         //     51,200
  float*    sess_lg = ws + 10400000;                         //     51,200

  const int T = 256;
  const int NE4  = NE / 4;                     // 2,500,000 float4
  const int gNE4 = (NE4 + T - 1) / T;
  const int gBE  = (BATCH * EMB + T - 1) / T;
  const int gG4  = (RM * (EMB / 4) + T - 1) / T;
  const int gBS  = (RM + T - 1) / T;
  const int gSP  = NNZ / 8;                    // 8 waves/block, 1 edge/wave

  // ---- HyperConv: item_hg = emb + L1 + L2 ----
  copy_f32_v4<<<gNE4, T, 0, stream>>>((float4*)item_hg, (const float4*)emb, NE4);
  fill_zero_f32_v4<<<gNE4, T, 0, stream>>>((float4*)h_b, NE4);
  spmm_coo<<<gSP, T, 0, stream>>>(adj_rows, adj_cols, adj_vals, emb, h_b);
  add_f32_v4<<<gNE4, T, 0, stream>>>((float4*)item_hg, (const float4*)h_b, NE4);
  fill_zero_f32_v4<<<gNE4, T, 0, stream>>>((float4*)h_a, NE4);
  spmm_coo<<<gSP, T, 0, stream>>>(adj_rows, adj_cols, adj_vals, h_b, h_a);
  add_f32_v4<<<gNE4, T, 0, stream>>>((float4*)item_hg, (const float4*)h_a, NE4);
  // h_a / h_b dead from here; post buffers may alias

  // ---- session embedding (hypergraph channel) ----
  gather_seq_v4<<<gG4, T, 0, stream>>>(rev, item_hg, seq_h);
  mean_seq<<<gBE, T, 0, stream>>>(seq_h, session_len, hs);
  hsg_kernel<<<gBE, T, 0, stream>>>(hs, glu2_w, hsg);

  // pack f16 operands (pad + transpose + convert)
  pack_X<<<(RM * K1P + T - 1) / T, T, 0, stream>>>(pos_w, seq_h, Xh);
  pack_w1t<<<(NP * K1P + T - 1) / T, T, 0, stream>>>(w1, W1t);
  pack_g1t<<<(NP * K2P + T - 1) / T, T, 0, stream>>>(glu1_w, G1t);
  fill_zero_f16<<<(RM * K2P + T - 1) / T, T, 0, stream>>>(nh0h, RM * K2P);

  gemm1_wmma<<<dim3(NP / 16, RM / 16), 32, 0, stream>>>(Xh, W1t, nh0h);
  gemm2_wmma<<<dim3(NP / 16, RM / 16), 32, 0, stream>>>(nh0h, G1t, glu1_b, hsg, nh);
  att_kernel<<<gBS, T, 0, stream>>>(nh, w2, mask, att);
  sesshg_kernel<<<BATCH, 128, 0, stream>>>(att, seq_h, sess_hg);

  // ---- LineConv channel ----
  s0_kernel<<<gBE, T, 0, stream>>>(session_item, emb, session_len, s_a, sess_lg);
  pack_f16<<<(BATCH * BATCH + T - 1) / T, T, 0, stream>>>(D, Dh, BATCH * BATCH);
  pack_t_f16<<<(BATCH * BATCH + T - 1) / T, T, 0, stream>>>(A, Ath);
  gemmDA_wmma<<<dim3(BATCH / 16, BATCH / 16), 32, 0, stream>>>(Dh, Ath, DA);
  sprop_kernel<<<gBE, T, 0, stream>>>(DA, s_a, s_b, sess_lg);
  sprop_kernel<<<gBE, T, 0, stream>>>(DA, s_b, s_a, sess_lg);

  // ---- SSL loss ----
  ssl_kernel<<<1, BATCH, 0, stream>>>(sess_hg, sess_lg, perm_rows, perm_cols, con);
}
